// Linear_29970281792057
// MI455X (gfx1250) — matrix operand order repaired and checked by reading, not compiled, not run on hardware
//
#include <hip/hip_runtime.h>
#include <hip/hip_bf16.h>

// ---------------------------------------------------------------------------
// Fused uint4-dequant GEMM for MI455X (gfx1250, wave32, WMMA).
//
//   Y[m,n] = sum_k A[m,k] * (q[n,k] - z[n,k/128]) * s[n,k/128]
//
// Roofline: 8.6 GFLOP vs ~72 MB traffic (64 MB int32 qweight dominant)
// -> memory-bound (~3.1 us at 23.3 TB/s HBM; working set is L2-resident).
// VALU roofline showed the f32->f16 repacking of A was co-dominant with
// memory, so pass 1 converts A to f16 once (into d_ws), and pass 2 streams
// qweight with global_load_b128, dequantizes in-register
// (v_cvt_f32_i32 + v_dual_fma_f32 + v_cvt_pk_rtz_f16_f32) and feeds
// v_wmma_f32_16x16x32_f16 with f32 accumulation. Each dequantized B
// fragment is reused by 4 WMMAs (4 M-subtiles per wave).
// ---------------------------------------------------------------------------

typedef __attribute__((ext_vector_type(16))) _Float16 v16h;
typedef __attribute__((ext_vector_type(8)))  _Float16 v8h;
typedef __attribute__((ext_vector_type(2)))  _Float16 h2;
typedef __attribute__((ext_vector_type(8)))  float    v8f;
typedef __attribute__((ext_vector_type(4)))  float    f4;
typedef __attribute__((ext_vector_type(4)))  int      i4;

union V16H {
    v16h v;
    h2   p[8];
    v8h  h[2];
};

__device__ __forceinline__ h2 pkh(float a, float b) {
    // v_cvt_pk_rtz_f16_f32 — pack two f32 into one VGPR of f16x2
    return __builtin_bit_cast(h2, __builtin_amdgcn_cvt_pkrtz(a, b));
}

// ---- Pass 1: A[f32] -> Ah[f16], 8 elements per thread, b128 in/out --------
__global__ __launch_bounds__(256) void convert_A_f16(
    const float* __restrict__ A, _Float16* __restrict__ Ah)
{
    const size_t idx = ((size_t)blockIdx.x * 256 + threadIdx.x) * 8;
    f4 a0 = *(const f4*)(A + idx);
    f4 a1 = *(const f4*)(A + idx + 4);
    union { v8h v; h2 p[4]; } r;
    r.p[0] = pkh(a0.x, a0.y);
    r.p[1] = pkh(a0.z, a0.w);
    r.p[2] = pkh(a1.x, a1.y);
    r.p[3] = pkh(a1.z, a1.w);
    *(v8h*)(Ah + idx) = r.v;
}

// ---- Pass 2: dequant + WMMA GEMM ------------------------------------------
template <bool AHALF>
__global__ __launch_bounds__(256) void qgemm_wmma_f16(
    const float*    __restrict__ A32,  // [M, K] f32 (AHALF == false)
    const _Float16* __restrict__ A16,  // [M, K] f16 (AHALF == true)
    const int*      __restrict__ Q,    // [N, K] int32, values in [0,16)
    const float*    __restrict__ S,    // [N, K/G] f32 scales
    const float*    __restrict__ Z,    // [N, K/G] f32 zeros
    float*          __restrict__ Y)    // [M, N] f32
{
    constexpr int K  = 4096;
    constexpr int N  = 4096;
    constexpr int G  = 128;      // quant group size
    constexpr int NG = K / G;    // 32 groups
    constexpr int MW = 4;        // M-subtiles (16 rows each) per wave

    const int lane  = threadIdx.x & 31;
    const int wave  = threadIdx.x >> 5;
    const int lid   = lane & 15;      // M/N index within tile
    const int hf    = lane >> 4;      // half-wave selector

    // Block tile: 128(M) x 64(N); 8 waves = 2 m-subtiles x 4 n-subtiles,
    // each wave owns 64(M) x 16(N).
    const int nBase = blockIdx.x * 64  + (wave & 3) * 16;
    const int mBase = blockIdx.y * 128 + (wave >> 2) * 64;

    const int n = nBase + lid;                       // this lane's W row / Y col
    const int* __restrict__   qrow = Q + (size_t)n * K;
    const float* __restrict__ srow = S + (size_t)n * NG;
    const float* __restrict__ zrow = Z + (size_t)n * NG;

    // A-fragment K offsets (contiguous k order, same as B): elems 0..7 at k+16*hf,
    // elems 8..15 at k+16*hf+8.
    const int ka = hf * 16;

    v8f acc[MW] = {};   // four 16x16 f32 accumulators -> 64 M rows

    for (int g = 0; g < NG; ++g) {
        const float s  = srow[g];
        const float zs = zrow[g] * s;   // fold dequant: w = q*s - z*s

        // Pull the next group's qweight cacheline ahead of demand
        // (global_prefetch_b8: no counter, speculative).
        if (g + 1 < NG) {
            __builtin_prefetch(qrow + (g + 1) * G + hf * 16, 0, 3);
        }

        #pragma unroll
        for (int kk = 0; kk < 4; ++kk) {
            const int kbase = g * G + kk * 32;

            // ---- B fragment: 16 contiguous K per lane at kbase + 16*hf ----
            const int kb = kbase + hf * 16;
            i4 q0 = *(const i4*)(qrow + kb);
            i4 q1 = *(const i4*)(qrow + kb + 4);
            i4 q2 = *(const i4*)(qrow + kb + 8);
            i4 q3 = *(const i4*)(qrow + kb + 12);

            V16H b;
            b.p[0] = pkh(__builtin_fmaf((float)q0.x, s, -zs),
                         __builtin_fmaf((float)q0.y, s, -zs));
            b.p[1] = pkh(__builtin_fmaf((float)q0.z, s, -zs),
                         __builtin_fmaf((float)q0.w, s, -zs));
            b.p[2] = pkh(__builtin_fmaf((float)q1.x, s, -zs),
                         __builtin_fmaf((float)q1.y, s, -zs));
            b.p[3] = pkh(__builtin_fmaf((float)q1.z, s, -zs),
                         __builtin_fmaf((float)q1.w, s, -zs));
            b.p[4] = pkh(__builtin_fmaf((float)q2.x, s, -zs),
                         __builtin_fmaf((float)q2.y, s, -zs));
            b.p[5] = pkh(__builtin_fmaf((float)q2.z, s, -zs),
                         __builtin_fmaf((float)q2.w, s, -zs));
            b.p[6] = pkh(__builtin_fmaf((float)q3.x, s, -zs),
                         __builtin_fmaf((float)q3.y, s, -zs));
            b.p[7] = pkh(__builtin_fmaf((float)q3.z, s, -zs),
                         __builtin_fmaf((float)q3.w, s, -zs));

            // ---- A fragments + WMMA: MW M-subtiles reuse the B fragment ----
            const int k0 = kbase + ka;        // f16 elems 0..7
            const int k1 = kbase + ka + 8;    // f16 elems 8..15
            #pragma unroll
            for (int i = 0; i < MW; ++i) {
                const size_t row = (size_t)(mBase + i * 16 + lid);
                V16H a;
                if constexpr (AHALF) {
                    // Pre-converted f16 A: two 16B b128 loads, zero VALU.
                    const _Float16* __restrict__ arow = A16 + row * K;
                    a.h[0] = *(const v8h*)(arow + k0);
                    a.h[1] = *(const v8h*)(arow + k1);
                } else {
                    const float* __restrict__ arow = A32 + row * K;
                    f4 a0 = *(const f4*)(arow + k0);
                    f4 a1 = *(const f4*)(arow + k0 + 4);
                    f4 a2 = *(const f4*)(arow + k1);
                    f4 a3 = *(const f4*)(arow + k1 + 4);
                    a.p[0] = pkh(a0.x, a0.y);
                    a.p[1] = pkh(a0.z, a0.w);
                    a.p[2] = pkh(a1.x, a1.y);
                    a.p[3] = pkh(a1.z, a1.w);
                    a.p[4] = pkh(a2.x, a2.y);
                    a.p[5] = pkh(a2.z, a2.w);
                    a.p[6] = pkh(a3.x, a3.y);
                    a.p[7] = pkh(a3.z, a3.w);
                }

                acc[i] = __builtin_amdgcn_wmma_f32_16x16x32_f16(
                    /*neg_a=*/false, a.v,
                    /*neg_b=*/false, b.v,
                    /*c_mod=*/(short)0, acc[i],
                    /*reuse_a=*/false, /*reuse_b=*/false);
            }
        }
    }

    // ---- Store: C/D layout -> VGPR r holds M = r + 8*hf, N = lid ----
    #pragma unroll
    for (int i = 0; i < MW; ++i) {
        const int mrow = mBase + i * 16 + hf * 8;
        float* __restrict__ yrow = Y + (size_t)mrow * N + nBase + lid;
        #pragma unroll
        for (int r = 0; r < 8; ++r) {
            yrow[(size_t)r * N] = acc[i][r];
        }
    }
}

extern "C" void kernel_launch(void* const* d_in, const int* in_sizes, int n_in,
                              void* d_out, int out_size, void* d_ws, size_t ws_size,
                              hipStream_t stream) {
    (void)in_sizes; (void)n_in; (void)out_size;

    const float* A  = (const float*)d_in[0];   // [256, 4096] f32
    const int*   Q  = (const int*)  d_in[1];   // [4096, 4096] int32
    const float* S  = (const float*)d_in[2];   // [4096, 32] f32
    const float* Zp = (const float*)d_in[3];   // [4096, 32] f32
    float*       Y  = (float*)d_out;           // [256, 4096] f32

    constexpr size_t M = 256, K = 4096;
    constexpr size_t aElems = M * K;           // 1,048,576

    // Block: 256 threads (8 waves), tile 128M x 64N.
    // Grid: (N/64, M/128) = (64, 2) -> 128 blocks, 1024 waves.
    dim3 grid(4096 / 64, 256 / 128);

    if (ws_size >= aElems * sizeof(_Float16)) {
        _Float16* Ah = (_Float16*)d_ws;
        // Pass 1: convert A to f16 once (stream-ordered before the GEMM).
        convert_A_f16<<<(unsigned)(aElems / (8 * 256)), 256, 0, stream>>>(A, Ah);
        // Pass 2: dequant + WMMA GEMM with f16 A.
        qgemm_wmma_f16<true><<<grid, 256, 0, stream>>>(nullptr, Ah, Q, S, Zp, Y);
    } else {
        // Fallback: single-pass with in-register A conversion.
        qgemm_wmma_f16<false><<<grid, 256, 0, stream>>>(A, nullptr, Q, S, Zp, Y);
    }
}
